// SwinTransformerBlock_64682207478499
// MI455X (gfx1250) — matrix operand order repaired and checked by reading, not compiled, not run on hardware
//
#include <hip/hip_runtime.h>
#include <hip/hip_bf16.h>
#include <math.h>

// ---------------------------------------------------------------------------
// Swin Transformer block for MI455X (gfx1250), wave32 + WMMA f16->f32,
// async global->LDS staging (GLOBAL_LOAD_ASYNC_TO_LDS_B128, ASYNCcnt).
// ---------------------------------------------------------------------------

typedef __attribute__((ext_vector_type(16))) _Float16 v16h;
typedef __attribute__((ext_vector_type(8)))  float    v8f;
typedef __attribute__((ext_vector_type(4)))  int      v4i;

union Frag {
    v16h     v;
    uint4    q[2];
    _Float16 h[16];
};

#define ACT_NONE 0
#define ACT_GELU 1

static constexpr int BB    = 32;
static constexpr int HH    = 56;
static constexpr int CC    = 256;
static constexpr int WS    = 7;
static constexpr int SHIFT = 3;
static constexpr int HEADS = 8;
static constexpr int HD    = 32;     // CC / HEADS
static constexpr int NSEQ  = 49;     // WS*WS
static constexpr int NW    = 64;     // (56/7)^2
static constexpr int NTOK  = BB * HH * HH;        // 100352 rows
static constexpr int HIDDEN = 4 * CC;             // 1024

// ---------------- async global->LDS staging (gfx1250 ASYNCcnt path) --------
#if defined(__has_builtin)
# if __has_builtin(__builtin_amdgcn_global_load_async_to_lds_b128)
#  define ASYNC_LDS 1
# endif
#endif
#ifndef ASYNC_LDS
# define ASYNC_LDS 0
#endif

typedef __attribute__((address_space(1))) v4i* gv4i_t;   // global int4*
typedef __attribute__((address_space(3))) v4i* sv4i_t;   // LDS    int4*

__device__ __forceinline__ void stage16(const _Float16* __restrict__ g,
                                        _Float16* __restrict__ l) {
#if ASYNC_LDS
    __builtin_amdgcn_global_load_async_to_lds_b128(
        (gv4i_t)(v4i*)g, (sv4i_t)(v4i*)l, 0, 0);
#else
    *(uint4*)l = *(const uint4*)g;
#endif
}

__device__ __forceinline__ void stage_wait() {
#if ASYNC_LDS
# if __has_builtin(__builtin_amdgcn_s_wait_asynccnt)
    __builtin_amdgcn_s_wait_asynccnt(0);
# else
    asm volatile("s_wait_asynccnt 0x0" ::: "memory");
# endif
#endif
}

// ---------------------------------------------------------------------------
// 0) weight convert f32 -> f16 WITH transpose: src [K][N] -> dst [N][K]
// ---------------------------------------------------------------------------
__global__ __launch_bounds__(256) void cvt_transpose(const float* __restrict__ s,
                                                     _Float16* __restrict__ d,
                                                     int K, int N) {
    int i = blockIdx.x * 256 + threadIdx.x;
    if (i < K * N) {
        int k = i / N, n = i % N;
        d[(size_t)n * K + k] = (_Float16)s[i];
    }
}

// ---------------------------------------------------------------------------
// 1) LayerNorm1 + roll(-SHIFT) + window partition -> xw (f16, [NTOK][256])
// ---------------------------------------------------------------------------
__global__ __launch_bounds__(256) void ln_shift_part(const float* __restrict__ x,
                                                     const float* __restrict__ g,
                                                     const float* __restrict__ b,
                                                     _Float16* __restrict__ xw) {
    const int row  = blockIdx.x * 8 + (threadIdx.x >> 5);
    const int lane = threadIdx.x & 31;

    const int bi  = row / (NW * NSEQ);
    const int rem = row % (NW * NSEQ);
    const int w   = rem / NSEQ;
    const int t   = rem % NSEQ;
    const int gh  = (w >> 3) * WS + t / WS;
    const int gw  = (w & 7)  * WS + t % WS;
    const int sh  = (gh + SHIFT) % HH;
    const int sw  = (gw + SHIFT) % HH;

    const float* xr = x + ((size_t)bi * (HH * HH) + (size_t)sh * HH + sw) * CC;

    float vals[8], s1 = 0.f, s2 = 0.f;
#pragma unroll
    for (int i = 0; i < 8; ++i) {
        float v = xr[i * 32 + lane];
        vals[i] = v; s1 += v; s2 += v * v;
    }
#pragma unroll
    for (int off = 16; off > 0; off >>= 1) {
        s1 += __shfl_xor(s1, off, 32);
        s2 += __shfl_xor(s2, off, 32);
    }
    const float mean = s1 * (1.f / CC);
    const float rstd = rsqrtf(s2 * (1.f / CC) - mean * mean + 1e-5f);

    _Float16* o = xw + (size_t)row * CC;
#pragma unroll
    for (int i = 0; i < 8; ++i) {
        int ch = i * 32 + lane;
        o[ch] = (_Float16)((vals[i] - mean) * rstd * g[ch] + b[ch]);
    }
}

// ---------------------------------------------------------------------------
// 2) Tiled WMMA GEMM: out[M,N] = act(A[M,K] * Bt[N,K]^T + bias) (+ resid)
// ---------------------------------------------------------------------------
template <int ACT, bool OUT_HALF, bool RESID>
__global__ __launch_bounds__(256) void gemm_wmma(const _Float16* __restrict__ A,
                                                 const _Float16* __restrict__ Bt,
                                                 const float* __restrict__ bias,
                                                 const float* __restrict__ resid,
                                                 void* __restrict__ out,
                                                 int M, int N, int K) {
    __shared__ _Float16 ldsA [64 * 48];   // [m][k], stride 48 halves (96B)
    __shared__ _Float16 ldsBt[64 * 48];   // [n][k], stride 48 halves

    const int tid  = threadIdx.x;
    const int lane = tid & 31;
    const int wave = tid >> 5;
    const int wm   = wave >> 2;           // 0..1 -> 32 rows
    const int wn   = wave & 3;            // 0..3 -> 16 cols
    const int nl   = lane & 15;
    const int hi   = lane >> 4;
    const int m0   = blockIdx.x * 64;
    const int n0   = blockIdx.y * 64;

    // tile loader: 64 rows x 32 cols, 8 halves (16B) per thread
    const int ar = tid >> 2;
    const int aq = (tid & 3) * 8;

    const _Float16* gA = A  + (size_t)(m0 + ar) * K + aq;
    const _Float16* gB = Bt + (size_t)(n0 + ar) * K + aq;
    _Float16* lA = &ldsA [ar * 48 + aq];
    _Float16* lB = &ldsBt[ar * 48 + aq];

    v8f acc[2] = {v8f{}, v8f{}};

    for (int kt = 0; kt < K; kt += 32) {
        __syncthreads();
        stage16(gA + kt, lA);
        stage16(gB + kt, lB);
        if (kt + 32 < K) {                       // warm GL2 for next k-tile
            __builtin_prefetch(gA + kt + 32, 0, 0);
            __builtin_prefetch(gB + kt + 32, 0, 0);
        }
        stage_wait();
        __syncthreads();

        Frag bf;
        {
            const _Float16* p = &ldsBt[(wn * 16 + nl) * 48];
            bf.q[0] = *(const uint4*)(p + hi * 8);
            bf.q[1] = *(const uint4*)(p + 16 + hi * 8);
        }
        Frag af0, af1;
        {
            const _Float16* p = &ldsA[(wm * 32 + nl) * 48];
            af0.q[0] = *(const uint4*)(p + hi * 8);
            af0.q[1] = *(const uint4*)(p + 16 + hi * 8);
            const _Float16* p1 = p + 16 * 48;
            af1.q[0] = *(const uint4*)(p1 + hi * 8);
            af1.q[1] = *(const uint4*)(p1 + 16 + hi * 8);
        }
        acc[0] = __builtin_amdgcn_wmma_f32_16x16x32_f16(false, af0.v, false, bf.v,
                                                        (short)0, acc[0], false, false);
        acc[1] = __builtin_amdgcn_wmma_f32_16x16x32_f16(false, af1.v, false, bf.v,
                                                        (short)0, acc[1], false, false);
    }

    const int   ncol = n0 + wn * 16 + nl;
    const float bv   = bias[ncol];
#pragma unroll
    for (int s = 0; s < 2; ++s) {
#pragma unroll
        for (int j = 0; j < 8; ++j) {
            int    m   = m0 + wm * 32 + s * 16 + j + 8 * hi;
            size_t idx = (size_t)m * N + ncol;
            float  v   = acc[s][j] + bv;
            if (ACT == ACT_GELU) v = 0.5f * v * (1.f + erff(v * 0.70710678f));
            if (RESID) v += resid[idx];
            if (OUT_HALF) ((_Float16*)out)[idx] = (_Float16)v;
            else          ((float*)out)[idx]    = v;
        }
    }
}

// ---------------------------------------------------------------------------
// 3) Windowed attention: one block (128 thr = 4 waves) per (window, head).
// ---------------------------------------------------------------------------
__device__ __forceinline__ int region_tag(int g) {
    return g < (HH - WS) ? 0 : (g < (HH - SHIFT) ? 1 : 2);
}

__global__ __launch_bounds__(128) void attn_win(const _Float16* __restrict__ qkv,
                                                const float* __restrict__ rel_bias,
                                                _Float16* __restrict__ outp) {
    __shared__ _Float16 qs[64 * 48];   // [m(64)][k(32)] padded, q * scale
    __shared__ _Float16 ks[64 * 48];   // [n(64)][k(32)]  (k row-major == Bt)
    __shared__ float    Sm[64 * 64];   // scores
    __shared__ _Float16 Pm[64 * 72];   // probs, [m(64)][k(64)] padded
    __shared__ _Float16 vs[32 * 72];   // [d(32)][s(64)]  (v transposed == Bt)
    __shared__ float    relS[169];     // rel_bias slice for this head
    __shared__ int      pinfo[NSEQ];   // r | c<<4 | regionTag<<8 per position

    const int w    = blockIdx.x >> 3;
    const int head = blockIdx.x & 7;
    const int tid  = threadIdx.x;
    const int lane = tid & 31;
    const int mt   = tid >> 5;
    const int nl   = lane & 15;
    const int hi   = lane >> 4;
    const int wh   = w >> 3, ww = w & 7;
    const float scale = 0.17677669529663689f;   // HD^-0.5

    const size_t qbase = (size_t)w * NSEQ * (3 * CC) + head * HD;

    // ---- stage q (scaled), k, v(transposed), bias slice, position table ----
    for (int i = tid; i < 169; i += 128) relS[i] = rel_bias[i * HEADS + head];
    if (tid < NSEQ) {
        int r = tid / WS, c = tid % WS;
        int tg = region_tag(wh * WS + r) * 3 + region_tag(ww * WS + c);
        pinfo[tid] = r | (c << 4) | (tg << 8);
    }
    {
        int r = tid >> 1, c0 = (tid & 1) * 16;          // 64 x 32
#pragma unroll
        for (int i = 0; i < 16; ++i) {
            float qv = 0.f, kv = 0.f;
            if (r < NSEQ) {
                qv = (float)qkv[qbase + (size_t)r * (3 * CC) + c0 + i];
                kv = (float)qkv[qbase + (size_t)r * (3 * CC) + CC + c0 + i];
            }
            qs[r * 48 + c0 + i] = (_Float16)(qv * scale);
            ks[r * 48 + c0 + i] = (_Float16)kv;
        }
        int d = tid >> 2, s0 = (tid & 3) * 16;          // 32 x 64 transposed
#pragma unroll
        for (int i = 0; i < 16; ++i) {
            int s = s0 + i;
            _Float16 vv = (_Float16)0.f;
            if (s < NSEQ) vv = qkv[qbase + (size_t)s * (3 * CC) + 2 * CC + d];
            vs[d * 72 + s] = vv;
        }
    }
    __syncthreads();

    // ---- S = q . kT  (fully branch-free bias/mask epilogue) ----
    Frag aq_;
    {
        const _Float16* p = &qs[(mt * 16 + nl) * 48];
        aq_.q[0] = *(const uint4*)(p + hi * 8);
        aq_.q[1] = *(const uint4*)(p + 16 + hi * 8);
    }
    // hoist the 8 row-position descriptors for this lane (clamped index)
    int   pmv[8];
    float mpen[8];
#pragma unroll
    for (int j = 0; j < 8; ++j) {
        int m   = mt * 16 + j + 8 * hi;
        mpen[j] = (m < NSEQ) ? 0.f : 1e30f;
        pmv[j]  = pinfo[(m < NSEQ) ? m : NSEQ - 1];
    }
#pragma unroll
    for (int nt = 0; nt < 4; ++nt) {
        Frag bk_;
        const _Float16* p = &ks[(nt * 16 + nl) * 48];
        bk_.q[0] = *(const uint4*)(p + hi * 8);
        bk_.q[1] = *(const uint4*)(p + 16 + hi * 8);
        v8f c = {};
        c = __builtin_amdgcn_wmma_f32_16x16x32_f16(false, aq_.v, false, bk_.v,
                                                   (short)0, c, false, false);
        const int   n    = nt * 16 + nl;
        const float npen = (n < NSEQ) ? 0.f : 1e30f;
        const int   pn   = pinfo[(n < NSEQ) ? n : NSEQ - 1];
#pragma unroll
        for (int j = 0; j < 8; ++j) {
            int m  = mt * 16 + j + 8 * hi;
            int pm = pmv[j];
            int ridx = ((pn & 15) - (pm & 15) + 6) * 13
                     + (((pn >> 4) & 15) - ((pm >> 4) & 15) + 6);
            float add = relS[ridx];                     // always executed
            add = ((pm >> 8) != (pn >> 8)) ? add - 100.f : add;
            // valid: c + bias + mask ; padded: ~ -1e30 (softmax -> 0)
            Sm[m * 64 + n] = c[j] + add - (mpen[j] + npen);
        }
    }
    __syncthreads();

    // ---- row softmax (one thread per row; rows >= 49 zeroed) ----
    if (tid < 64) {
        if (tid < NSEQ) {
            float mx = -1e30f;
            for (int n = 0; n < 64; ++n) mx = fmaxf(mx, Sm[tid * 64 + n]);
            float sum = 0.f;
            for (int n = 0; n < 64; ++n) {
                float e = __expf(Sm[tid * 64 + n] - mx);
                Sm[tid * 64 + n] = e; sum += e;
            }
            float inv = 1.f / sum;
            for (int n = 0; n < 64; ++n)
                Pm[tid * 72 + n] = (_Float16)(Sm[tid * 64 + n] * inv);
        } else {
            for (int n = 0; n < 64; ++n) Pm[tid * 72 + n] = (_Float16)0.f;
        }
    }
    __syncthreads();

    // ---- O = P . v ----
#pragma unroll
    for (int nt = 0; nt < 2; ++nt) {
        v8f c = {};
#pragma unroll
        for (int kt = 0; kt < 2; ++kt) {
            Frag ap, bv_;
            const _Float16* pa = &Pm[(mt * 16 + nl) * 72 + kt * 32];
            ap.q[0] = *(const uint4*)(pa + hi * 8);
            ap.q[1] = *(const uint4*)(pa + 16 + hi * 8);
            const _Float16* pb = &vs[(nt * 16 + nl) * 72 + kt * 32];
            bv_.q[0] = *(const uint4*)(pb + hi * 8);
            bv_.q[1] = *(const uint4*)(pb + 16 + hi * 8);
            c = __builtin_amdgcn_wmma_f32_16x16x32_f16(false, ap.v, false, bv_.v,
                                                       (short)0, c, false, false);
        }
#pragma unroll
        for (int j = 0; j < 8; ++j) {
            int m = mt * 16 + j + 8 * hi;
            if (m < NSEQ) {
                int d = nt * 16 + nl;
                outp[((size_t)w * NSEQ + m) * CC + head * HD + d] = (_Float16)c[j];
            }
        }
    }
}

// ---------------------------------------------------------------------------
// 4) window reverse + roll(+SHIFT) + residual + LayerNorm2
// ---------------------------------------------------------------------------
__global__ __launch_bounds__(256) void resid_ln2(const float* __restrict__ x,
                                                 const _Float16* __restrict__ proj,
                                                 const float* __restrict__ g,
                                                 const float* __restrict__ b,
                                                 float* __restrict__ x2,
                                                 _Float16* __restrict__ h2) {
    const int row  = blockIdx.x * 8 + (threadIdx.x >> 5);
    const int lane = threadIdx.x & 31;

    const int bi = row / (HH * HH);
    const int hw = row % (HH * HH);
    const int hh = hw / HH, wwp = hw % HH;
    const int gh = (hh + HH - SHIFT) % HH;
    const int gw = (wwp + HH - SHIFT) % HH;
    const size_t wrow = ((size_t)bi * NW + (gh / WS) * 8 + gw / WS) * NSEQ
                      + (gh % WS) * WS + (gw % WS);

    const float*    xr = x    + (size_t)row * CC;
    const _Float16* pr = proj + wrow * CC;

    float vals[8], s1 = 0.f, s2 = 0.f;
#pragma unroll
    for (int i = 0; i < 8; ++i) {
        int ch = i * 32 + lane;
        float v = xr[ch] + (float)pr[ch];
        vals[i] = v; s1 += v; s2 += v * v;
    }
#pragma unroll
    for (int off = 16; off > 0; off >>= 1) {
        s1 += __shfl_xor(s1, off, 32);
        s2 += __shfl_xor(s2, off, 32);
    }
    const float mean = s1 * (1.f / CC);
    const float rstd = rsqrtf(s2 * (1.f / CC) - mean * mean + 1e-5f);

    float*    xo = x2 + (size_t)row * CC;
    _Float16* ho = h2 + (size_t)row * CC;
#pragma unroll
    for (int i = 0; i < 8; ++i) {
        int ch = i * 32 + lane;
        xo[ch] = vals[i];
        ho[ch] = (_Float16)((vals[i] - mean) * rstd * g[ch] + b[ch]);
    }
}

// ---------------------------------------------------------------------------
// host launcher
// ---------------------------------------------------------------------------
extern "C" void kernel_launch(void* const* d_in, const int* in_sizes, int n_in,
                              void* d_out, int out_size, void* d_ws, size_t ws_size,
                              hipStream_t stream) {
    (void)in_sizes; (void)n_in; (void)out_size; (void)ws_size;

    const float* x      = (const float*)d_in[0];
    const float* n1g    = (const float*)d_in[1];
    const float* n1b    = (const float*)d_in[2];
    const float* qkv_w  = (const float*)d_in[3];
    const float* qkv_b  = (const float*)d_in[4];
    const float* relb   = (const float*)d_in[5];
    const float* proj_w = (const float*)d_in[6];
    const float* proj_b = (const float*)d_in[7];
    const float* n2g    = (const float*)d_in[8];
    const float* n2b    = (const float*)d_in[9];
    const float* fc1_w  = (const float*)d_in[10];
    const float* fc1_b  = (const float*)d_in[11];
    const float* fc2_w  = (const float*)d_in[12];
    const float* fc2_b  = (const float*)d_in[13];

    char*  ws  = (char*)d_ws;
    size_t off = 0;
    auto alloc = [&](size_t bytes) -> char* {
        char* p = ws + off;
        off += (bytes + 255) & ~(size_t)255;
        return p;
    };

    const size_t tokC_h = (size_t)NTOK * CC * sizeof(_Float16);
    const size_t tokC_f = (size_t)NTOK * CC * sizeof(float);
    const size_t qkv_sz = (size_t)NTOK * 3 * CC * sizeof(_Float16);

    _Float16* wq = (_Float16*)alloc((size_t)CC * 3 * CC * 2);   // [768][256]
    _Float16* wp = (_Float16*)alloc((size_t)CC * CC * 2);       // [256][256]
    _Float16* w1 = (_Float16*)alloc((size_t)CC * HIDDEN * 2);   // [1024][256]
    _Float16* w2 = (_Float16*)alloc((size_t)HIDDEN * CC * 2);   // [256][1024]

    char* regA = alloc(tokC_h + qkv_sz);          // [xw | qkv] -> reused as hid
    _Float16* xw   = (_Float16*)regA;
    _Float16* qkvb = (_Float16*)(regA + tokC_h);
    _Float16* hid  = (_Float16*)regA;

    char* regB = alloc(tokC_h);                   // attn out -> reused as h2
    _Float16* attnb = (_Float16*)regB;
    _Float16* h2    = (_Float16*)regB;

    _Float16* projo = (_Float16*)alloc(tokC_h);
    float*    x2    = (float*)alloc(tokC_f);
    float*    out   = (float*)d_out;

    // weight conversion + transpose to [N][K]
    cvt_transpose<<<(CC * 3 * CC + 255) / 256, 256, 0, stream>>>(qkv_w, wq, CC, 3 * CC);
    cvt_transpose<<<(CC * CC + 255) / 256, 256, 0, stream>>>(proj_w, wp, CC, CC);
    cvt_transpose<<<(CC * HIDDEN + 255) / 256, 256, 0, stream>>>(fc1_w, w1, CC, HIDDEN);
    cvt_transpose<<<(HIDDEN * CC + 255) / 256, 256, 0, stream>>>(fc2_w, w2, HIDDEN, CC);

    // LN1 + shift + partition
    ln_shift_part<<<NTOK / 8, 256, 0, stream>>>(x, n1g, n1b, xw);

    // QKV gemm
    gemm_wmma<ACT_NONE, true, false><<<dim3(NTOK / 64, (3 * CC) / 64), 256, 0, stream>>>(
        xw, wq, qkv_b, nullptr, qkvb, NTOK, 3 * CC, CC);

    // attention
    attn_win<<<BB * NW * HEADS, 128, 0, stream>>>(qkvb, relb, attnb);

    // proj gemm
    gemm_wmma<ACT_NONE, true, false><<<dim3(NTOK / 64, CC / 64), 256, 0, stream>>>(
        attnb, wp, proj_b, nullptr, projo, NTOK, CC, CC);

    // reverse + residual + LN2
    resid_ln2<<<NTOK / 8, 256, 0, stream>>>(x, projo, n2g, n2b, x2, h2);

    // fc1 gemm + exact GELU
    gemm_wmma<ACT_GELU, true, false><<<dim3(NTOK / 64, HIDDEN / 64), 256, 0, stream>>>(
        h2, w1, fc1_b, nullptr, hid, NTOK, HIDDEN, CC);

    // fc2 gemm + residual -> f32 out
    gemm_wmma<ACT_NONE, false, true><<<dim3(NTOK / 64, CC / 64), 256, 0, stream>>>(
        hid, w2, fc2_b, x2, out, NTOK, CC, HIDDEN);
}
